// NeuralMemory_18348100288794
// MI455X (gfx1250) — compile-verified, matrix operand order checked by reading, not run on hardware
//
#include <hip/hip_runtime.h>

// MI455X / gfx1250, wave32. bf16 WMMA 16x16x32 with fp32 accumulate.
// Sequential TTT scan: 64 chunks x 6 kernels, state fp32 in global (L2-resident),
// bf16 mirrors for the tensor-core path. LDS staging uses the CDNA5 async
// global->LDS direct path (GLOBAL_LOAD_ASYNC_TO_LDS_B128 + s_wait_asynccnt).
// ~94MB workspace.

typedef __attribute__((ext_vector_type(16))) __bf16 v16bf;
typedef __attribute__((ext_vector_type(8)))  float  v8f;

#define BB  4
#define SS  4096
#define DD  512
#define HH  512
#define CC  64
#define NCH 64

struct P {
  const float *x,*Wq,*Wk,*Wv,*init_w1,*init_b1,*init_w2,*init_b2;
  const float *alpha_w,*alpha_b,*eta_w,*eta_b,*theta_w,*theta_b;
  __bf16 *Wq_bf,*Wk_bf,*Wv_bf;
  __bf16 *qn,*kn,*knT,*vbf;                 // qn/kn/v: [B][S][D]; knT: [B][NC][D][C]
  float  *w1,*mw1,*w2,*mw2,*b1,*mb1,*b2,*mb2;
  __bf16 *w1_bf,*w2_bf,*w2T_bf;             // bf16 mirrors (w1:[H][D], w2:[D][H], w2T:[H][D])
  float  *gates;                            // [3][B][NC] : a,e,t
  __bf16 *hq,*hk,*hkT;  float *gate;        // per-chunk intermediates
  __bf16 *dvp,*dvpT,*dhT;
  float  *out;
};

__device__ __forceinline__ float sigm(float x){ return 1.0f/(1.0f + __expf(-x)); }

__device__ __forceinline__ v8f v8zero() {
  v8f z; for (int i=0;i<8;++i) z[i]=0.f; return z;
}

__device__ __forceinline__ v8f wmma_bf(v16bf a, v16bf b, v8f c) {
  return __builtin_amdgcn_wmma_f32_16x16x32_bf16(false, a, false, b, (short)0, c, false, false);
}

// Load a 16x32 bf16 fragment (A layout; B operand loads B^T with same layout).
// Per-lane: row = lane&15; halves 0..7 at K = (lane>>4)*8, halves 8..15 at K = 16+(lane>>4)*8.
__device__ __forceinline__ v16bf ldfrag(const __bf16* base, int ld, int lane) {
  const __bf16* q = base + (size_t)(lane & 15) * ld + ((lane >> 4) << 3);
  v16bf r;
  ((uint4*)&r)[0] = *(const uint4*)q;
  ((uint4*)&r)[1] = *(const uint4*)(q + 16);
  return r;
}

// CDNA5 async direct global->LDS copy (16B per lane), ASYNCcnt-tracked.
// LDS byte offset = low 32 bits of the generic address of a __shared__ object
// (flat aperture places LDS offsets in addr[31:0]).
__device__ __forceinline__ void async_cp16(void* lds_dst, const void* gsrc) {
  unsigned off = (unsigned)(unsigned long long)lds_dst;
  asm volatile("global_load_async_to_lds_b128 %0, %1, off"
               :: "v"(off), "v"(gsrc) : "memory");
}
__device__ __forceinline__ void async_wait0() {
  asm volatile("s_wait_asynccnt 0x0" ::: "memory");
}

// ---------------- setup kernels ----------------

__global__ __launch_bounds__(256) void k_cvt(P p) {
  int i = blockIdx.x*256 + threadIdx.x;
  if (i < DD*DD) {
    p.Wq_bf[i] = (__bf16)p.Wq[i];
    p.Wk_bf[i] = (__bf16)p.Wk[i];
    p.Wv_bf[i] = (__bf16)p.Wv[i];
  }
}

__global__ __launch_bounds__(256) void k_init(P p) {
  int b = blockIdx.y;
  int i = blockIdx.x*256 + threadIdx.x;
  if (i >= HH*DD) return;
  size_t o = (size_t)b*HH*DD + i;
  float w1v = p.init_w1[i];
  p.w1[o]=w1v; p.mw1[o]=0.f; p.w1_bf[o]=(__bf16)w1v;
  float w2v = p.init_w2[i];                 // i = d*H + h
  p.w2[o]=w2v; p.mw2[o]=0.f; p.w2_bf[o]=(__bf16)w2v;
  int d = i >> 9, h = i & 511;
  p.w2T_bf[(size_t)b*HH*DD + (size_t)h*DD + d] = (__bf16)w2v;
  if (i < HH) {
    p.b1[b*HH+i]=p.init_b1[i]; p.mb1[b*HH+i]=0.f;
    p.b2[b*DD+i]=p.init_b2[i]; p.mb2[b*DD+i]=0.f;
  }
}

__global__ __launch_bounds__(64) void k_gates(P p) {
  int b  = blockIdx.x >> 6;
  int ci = blockIdx.x & 63;
  int t  = threadIdx.x;                     // 0..63 token in chunk
  const float* xr = p.x + (((size_t)b*SS) + (size_t)ci*CC + t) * DD;
  float sa=0.f, se=0.f, st=0.f;
  for (int j=0;j<DD;j+=4) {
    float4 xv = *(const float4*)(xr+j);
    float4 aw = *(const float4*)(p.alpha_w+j);
    float4 ew = *(const float4*)(p.eta_w+j);
    float4 tw = *(const float4*)(p.theta_w+j);
    sa += xv.x*aw.x + xv.y*aw.y + xv.z*aw.z + xv.w*aw.w;
    se += xv.x*ew.x + xv.y*ew.y + xv.z*ew.z + xv.w*ew.w;
    st += xv.x*tw.x + xv.y*tw.y + xv.z*tw.z + xv.w*tw.w;
  }
  sa = sigm(sa + p.alpha_b[0]);
  se = sigm(se + p.eta_b[0]);
  st = sigm(st + p.theta_b[0]);
  __shared__ float ra[CC], re[CC], rt[CC];
  ra[t]=sa; re[t]=se; rt[t]=st;
  __syncthreads();
  if (t==0) {
    float A=0.f,E=0.f,T=0.f;
    for (int j=0;j<CC;++j){ A+=ra[j]; E+=re[j]; T+=rt[j]; }
    p.gates[(0*BB+b)*NCH+ci]=A*(1.f/CC);
    p.gates[(1*BB+b)*NCH+ci]=E*(1.f/CC);
    p.gates[(2*BB+b)*NCH+ci]=T*(1.f/CC);
  }
}

// q,k,v projections + fused row L2-normalize. Block = 16-token strip, 8 waves.
__global__ __launch_bounds__(256) void k_qkvn(P p) {
  __shared__ __bf16 xs[16*DD];
  __shared__ float  rowsq[2][16];
  int strip = blockIdx.x;
  int b  = strip >> 8;                      // 256 strips per batch
  int s0 = (strip & 255) * 16;
  int tid = threadIdx.x, lane = tid & 31, w = tid >> 5;
  const float* xg = p.x + ((size_t)b*SS + s0) * DD;
  for (int i = tid; i < 16*DD/4; i += 256) {
    float4 xv = *(const float4*)(xg + (size_t)i*4);
    __bf16* dst = xs + i*4;
    dst[0]=(__bf16)xv.x; dst[1]=(__bf16)xv.y; dst[2]=(__bf16)xv.z; dst[3]=(__bf16)xv.w;
  }
  if (tid < 32) (&rowsq[0][0])[tid] = 0.f;
  __syncthreads();
  v8f aq[4], ak[4], av[4];
  for (int j=0;j<4;++j){ aq[j]=v8zero(); ak[j]=v8zero(); av[j]=v8zero(); }
  for (int k0=0;k0<DD;k0+=32) {
    v16bf af = ldfrag(xs + k0, DD, lane);
    for (int j=0;j<4;++j) {
      int n0 = (w + 8*j) * 16;
      v16bf bq = ldfrag(p.Wq_bf + (size_t)n0*DD + k0, DD, lane);
      v16bf bk = ldfrag(p.Wk_bf + (size_t)n0*DD + k0, DD, lane);
      v16bf bv = ldfrag(p.Wv_bf + (size_t)n0*DD + k0, DD, lane);
      aq[j] = wmma_bf(af,bq,aq[j]);
      ak[j] = wmma_bf(af,bk,ak[j]);
      av[j] = wmma_bf(af,bv,av[j]);
    }
  }
  int hi = lane >> 4, ln = lane & 15;
  float pq[8], pk[8];
  for (int g=0; g<8; ++g){ pq[g]=0.f; pk[g]=0.f; }
  for (int j=0;j<4;++j) {
    int n = (w + 8*j)*16 + ln;
    for (int g=0; g<8; ++g) {
      int m = g + 8*hi;
      p.vbf[((size_t)b*SS + s0 + m)*DD + n] = (__bf16)av[j][g];
      float qv = aq[j][g], kv = ak[j][g];
      pq[g] += qv*qv; pk[g] += kv*kv;
    }
  }
  for (int g=0; g<8; ++g) {
    atomicAdd(&rowsq[0][g + 8*hi], pq[g]);
    atomicAdd(&rowsq[1][g + 8*hi], pk[g]);
  }
  __syncthreads();
  int chunk = s0 >> 6;
  int cbase = s0 & 63;
  for (int j=0;j<4;++j) {
    int n = (w + 8*j)*16 + ln;
    __bf16 ktmp[8];
    for (int g=0; g<8; ++g) {
      int m = g + 8*hi;
      float scq = 1.f / fmaxf(sqrtf(rowsq[0][m]), 1e-12f);
      float sck = 1.f / fmaxf(sqrtf(rowsq[1][m]), 1e-12f);
      float qv = aq[j][g]*scq, kv = ak[j][g]*sck;
      p.qn[((size_t)b*SS + s0 + m)*DD + n] = (__bf16)qv;
      p.kn[((size_t)b*SS + s0 + m)*DD + n] = (__bf16)kv;
      ktmp[g] = (__bf16)kv;
    }
    // transposed store: contiguous 8 elems along token dim
    *(uint4*)(p.knT + (((size_t)b*NCH + chunk)*DD + n)*CC + cbase + 8*hi) = *(uint4*)ktmp;
  }
}

// ---------------- per-chunk scan kernels ----------------
// grid B*16, each block: one 16-row c-strip, 8 waves cover 8 of 32 n-tiles x 4 t-blocks.

__global__ __launch_bounds__(256) void kA(P p, int ci) {   // pre/hq/hk/gate
  __shared__ __bf16 qs[16*DD], ks[16*DD];
  int blk = blockIdx.x; int b = blk >> 4, t = blk & 15;
  int tid=threadIdx.x, lane=tid&31, w=tid>>5;
  int ctile = t >> 2;
  int h0 = ((t & 3)*8 + w) * 16;
  size_t tok0 = (size_t)b*SS + (size_t)ci*CC + ctile*16;
  const uint4* qg=(const uint4*)(p.qn + tok0*DD);
  const uint4* kg=(const uint4*)(p.kn + tok0*DD);
  uint4* qsv=(uint4*)qs; uint4* ksv=(uint4*)ks;
  for (int i=tid;i<16*DD/8;i+=256){ async_cp16(qsv+i, qg+i); async_cp16(ksv+i, kg+i); }
  async_wait0();
  __syncthreads();
  v8f accq=v8zero(), acck=v8zero();
  const __bf16* w1p = p.w1_bf + ((size_t)b*HH + h0)*DD;
  for (int k0=0;k0<DD;k0+=32) {
    v16bf bw  = ldfrag(w1p + k0, DD, lane);
    v16bf aqf = ldfrag(qs + k0, DD, lane);
    v16bf akf = ldfrag(ks + k0, DD, lane);
    accq = wmma_bf(aqf,bw,accq);
    acck = wmma_bf(akf,bw,acck);
  }
  int hi=lane>>4, ln=lane&15;
  int h = h0 + ln;
  float b1h = p.b1[b*HH + h];
  __bf16 hkT_t[8];
  for (int g=0; g<8; ++g) {
    int c = ctile*16 + 8*hi + g;
    float pre = acck[g] + b1h;
    float sg  = sigm(pre);
    float hkv = pre*sg;
    float gt  = sg*(1.f + pre*(1.f - sg));
    float zq  = accq[g] + b1h;
    float hqv = zq*sigm(zq);
    p.hq  [((size_t)b*CC + c)*HH + h] = (__bf16)hqv;
    p.hk  [((size_t)b*CC + c)*HH + h] = (__bf16)hkv;
    p.gate[((size_t)b*CC + c)*HH + h] = gt;
    hkT_t[g] = (__bf16)hkv;
  }
  *(uint4*)(p.hkT + ((size_t)b*HH + h)*CC + ctile*16 + 8*hi) = *(uint4*)hkT_t;
}

__global__ __launch_bounds__(256) void kB(P p, int ci) {   // y (out), vp, dvp
  __shared__ __bf16 hqs[16*HH], hks[16*HH];
  int blk=blockIdx.x; int b=blk>>4, t=blk&15;
  int tid=threadIdx.x, lane=tid&31, w=tid>>5;
  int ctile=t>>2;
  int d0=((t&3)*8+w)*16;
  const uint4* ag=(const uint4*)(p.hq + ((size_t)b*CC + ctile*16)*HH);
  const uint4* bg=(const uint4*)(p.hk + ((size_t)b*CC + ctile*16)*HH);
  uint4* s1=(uint4*)hqs; uint4* s2=(uint4*)hks;
  for (int i=tid;i<16*HH/8;i+=256){ async_cp16(s1+i, ag+i); async_cp16(s2+i, bg+i); }
  async_wait0();
  __syncthreads();
  v8f accy=v8zero(), accv=v8zero();
  const __bf16* w2p = p.w2_bf + ((size_t)b*DD + d0)*HH;
  for (int k0=0;k0<HH;k0+=32) {
    v16bf bw = ldfrag(w2p + k0, HH, lane);
    v16bf a1 = ldfrag(hqs + k0, HH, lane);
    v16bf a2 = ldfrag(hks + k0, HH, lane);
    accy = wmma_bf(a1,bw,accy);
    accv = wmma_bf(a2,bw,accv);
  }
  int hi=lane>>4, ln=lane&15;
  int d = d0 + ln;
  float b2d = p.b2[b*DD + d];
  __bf16 dvT_t[8];
  for (int g=0; g<8; ++g) {
    int c = ctile*16 + 8*hi + g;
    size_t tok = (size_t)b*SS + (size_t)ci*CC + c;
    p.out[tok*DD + d] = accy[g] + b2d;
    float vp = accv[g] + b2d;
    float vv = (float)p.vbf[tok*DD + d];
    float dv = (vp - vv) * (2.f/DD);
    p.dvp[((size_t)b*CC + c)*DD + d] = (__bf16)dv;
    dvT_t[g] = (__bf16)dv;
  }
  *(uint4*)(p.dvpT + ((size_t)b*DD + d)*CC + ctile*16 + 8*hi) = *(uint4*)dvT_t;
}

__global__ __launch_bounds__(256) void kC1(P p, int ci) {  // dh = (dvp @ w2) * gate
  __shared__ __bf16 dsv[16*DD];
  int blk=blockIdx.x; int b=blk>>4, t=blk&15;
  int tid=threadIdx.x, lane=tid&31, w=tid>>5;
  int ctile=t>>2;
  int h0=((t&3)*8+w)*16;
  const uint4* ag=(const uint4*)(p.dvp + ((size_t)b*CC + ctile*16)*DD);
  uint4* sv=(uint4*)dsv;
  for (int i=tid;i<16*DD/8;i+=256) async_cp16(sv+i, ag+i);
  async_wait0();
  __syncthreads();
  v8f acc=v8zero();
  const __bf16* w2tp = p.w2T_bf + ((size_t)b*HH + h0)*DD;
  for (int k0=0;k0<DD;k0+=32) {
    v16bf bw = ldfrag(w2tp + k0, DD, lane);
    v16bf a  = ldfrag(dsv + k0, DD, lane);
    acc = wmma_bf(a,bw,acc);
  }
  int hi=lane>>4, ln=lane&15; int h=h0+ln;
  __bf16 dhT_t[8];
  for (int g=0; g<8; ++g) {
    int c = ctile*16 + 8*hi + g;
    float gt = p.gate[((size_t)b*CC + c)*HH + h];
    dhT_t[g] = (__bf16)(acc[g]*gt);
  }
  *(uint4*)(p.dhT + ((size_t)b*HH + h)*CC + ctile*16 + 8*hi) = *(uint4*)dhT_t;
}

__global__ __launch_bounds__(256) void kC2(P p, int ci) {  // g_w2 + fused w2/mw2 update
  __shared__ __bf16 asv[16*CC];
  int blk=blockIdx.x; int b=blk>>7, t=blk&127;
  int tid=threadIdx.x, lane=tid&31, w=tid>>5;
  int dtile=t>>2; int d0=dtile*16;
  int h0=((t&3)*8+w)*16;
  const uint4* ag=(const uint4*)(p.dvpT + ((size_t)b*DD + d0)*CC);
  uint4* sv=(uint4*)asv;
  for (int i=tid;i<16*CC/8;i+=256) async_cp16(sv+i, ag+i);
  async_wait0();
  __syncthreads();
  v8f acc=v8zero();
  for (int k0=0;k0<CC;k0+=32) {
    v16bf a  = ldfrag(asv + k0, CC, lane);
    v16bf bb = ldfrag(p.hkT + ((size_t)b*HH + h0)*CC + k0, CC, lane);
    acc = wmma_bf(a,bb,acc);
  }
  int hi=lane>>4, ln=lane&15; int h=h0+ln;
  float av = p.gates[(0*BB+b)*NCH+ci];
  float ev = p.gates[(1*BB+b)*NCH+ci];
  float tv = p.gates[(2*BB+b)*NCH+ci];
  __bf16 wT_t[8];
  for (int g=0; g<8; ++g) {
    int d = d0 + 8*hi + g;
    size_t o = ((size_t)b*DD + d)*HH + h;
    float gw = acc[g]*(1.f/CC);
    float nm = ev*p.mw2[o] - tv*gw;
    float nw = (1.f-av)*p.w2[o] + nm;
    p.mw2[o]=nm; p.w2[o]=nw;
    p.w2_bf[o]=(__bf16)nw;
    wT_t[g]=(__bf16)nw;
  }
  *(uint4*)(p.w2T_bf + ((size_t)b*HH + h)*DD + d0 + 8*hi) = *(uint4*)wT_t;
}

__global__ __launch_bounds__(256) void kD(P p, int ci) {   // g_w1 + fused w1/mw1 update
  __shared__ __bf16 asv[16*CC];
  int blk=blockIdx.x; int b=blk>>7, t=blk&127;
  int tid=threadIdx.x, lane=tid&31, w=tid>>5;
  int htile=t>>2; int hm0=htile*16;
  int d0=((t&3)*8+w)*16;
  const uint4* ag=(const uint4*)(p.dhT + ((size_t)b*HH + hm0)*CC);
  uint4* sv=(uint4*)asv;
  for (int i=tid;i<16*CC/8;i+=256) async_cp16(sv+i, ag+i);
  async_wait0();
  __syncthreads();
  v8f acc=v8zero();
  const __bf16* knTp = p.knT + (((size_t)b*NCH + ci)*DD + d0)*CC;
  for (int k0=0;k0<CC;k0+=32) {
    v16bf a  = ldfrag(asv + k0, CC, lane);
    v16bf bb = ldfrag(knTp + k0, CC, lane);
    acc = wmma_bf(a,bb,acc);
  }
  int hi=lane>>4, ln=lane&15; int d=d0+ln;
  float av = p.gates[(0*BB+b)*NCH+ci];
  float ev = p.gates[(1*BB+b)*NCH+ci];
  float tv = p.gates[(2*BB+b)*NCH+ci];
  for (int g=0; g<8; ++g) {
    int h = hm0 + 8*hi + g;
    size_t o = ((size_t)b*HH + h)*DD + d;
    float gw = acc[g]*(1.f/CC);
    float nm = ev*p.mw1[o] - tv*gw;
    float nw = (1.f-av)*p.w1[o] + nm;
    p.mw1[o]=nm; p.w1[o]=nw;
    p.w1_bf[o]=(__bf16)nw;
  }
}

__global__ __launch_bounds__(512) void kUb(P p, int ci) {  // b1/b2 momentum updates
  int b=blockIdx.x, i=threadIdx.x;
  float av=p.gates[(0*BB+b)*NCH+ci];
  float ev=p.gates[(1*BB+b)*NCH+ci];
  float tv=p.gates[(2*BB+b)*NCH+ci];
  const __bf16* r1 = p.dhT  + ((size_t)b*HH + i)*CC;
  const __bf16* r2 = p.dvpT + ((size_t)b*DD + i)*CC;
  float s1=0.f, s2=0.f;
  for (int c=0;c<CC;++c){ s1 += (float)r1[c]; s2 += (float)r2[c]; }
  int o=b*HH+i;
  float nm1 = ev*p.mb1[o] - tv*(s1*(1.f/CC));
  p.mb1[o]=nm1; p.b1[o]=(1.f-av)*p.b1[o] + nm1;
  float nm2 = ev*p.mb2[o] - tv*(s2*(1.f/CC));
  p.mb2[o]=nm2; p.b2[o]=(1.f-av)*p.b2[o] + nm2;
}

// ---------------- host launcher ----------------

extern "C" void kernel_launch(void* const* d_in, const int* in_sizes, int n_in,
                              void* d_out, int out_size, void* d_ws, size_t ws_size,
                              hipStream_t stream) {
  P p;
  p.x       = (const float*)d_in[0];
  p.Wq      = (const float*)d_in[1];
  p.Wk      = (const float*)d_in[2];
  p.Wv      = (const float*)d_in[3];
  p.init_w1 = (const float*)d_in[4];
  p.init_b1 = (const float*)d_in[5];
  p.init_w2 = (const float*)d_in[6];
  p.init_b2 = (const float*)d_in[7];
  p.alpha_w = (const float*)d_in[8];
  p.alpha_b = (const float*)d_in[9];
  p.eta_w   = (const float*)d_in[10];
  p.eta_b   = (const float*)d_in[11];
  p.theta_w = (const float*)d_in[12];
  p.theta_b = (const float*)d_in[13];
  p.out     = (float*)d_out;

  char* ws = (char*)d_ws;
  size_t off = 0;
  auto take = [&](size_t elems, size_t esz)->void* {
    void* q = ws + off;
    off += (elems*esz + 255) & ~(size_t)255;
    return q;
  };
  p.Wq_bf  = (__bf16*)take((size_t)DD*DD, 2);
  p.Wk_bf  = (__bf16*)take((size_t)DD*DD, 2);
  p.Wv_bf  = (__bf16*)take((size_t)DD*DD, 2);
  p.qn     = (__bf16*)take((size_t)BB*SS*DD, 2);
  p.kn     = (__bf16*)take((size_t)BB*SS*DD, 2);
  p.knT    = (__bf16*)take((size_t)BB*NCH*DD*CC, 2);
  p.vbf    = (__bf16*)take((size_t)BB*SS*DD, 2);
  p.w1     = (float*)take((size_t)BB*HH*DD, 4);
  p.mw1    = (float*)take((size_t)BB*HH*DD, 4);
  p.w2     = (float*)take((size_t)BB*DD*HH, 4);
  p.mw2    = (float*)take((size_t)BB*DD*HH, 4);
  p.b1     = (float*)take((size_t)BB*HH, 4);
  p.mb1    = (float*)take((size_t)BB*HH, 4);
  p.b2     = (float*)take((size_t)BB*DD, 4);
  p.mb2    = (float*)take((size_t)BB*DD, 4);
  p.w1_bf  = (__bf16*)take((size_t)BB*HH*DD, 2);
  p.w2_bf  = (__bf16*)take((size_t)BB*DD*HH, 2);
  p.w2T_bf = (__bf16*)take((size_t)BB*HH*DD, 2);
  p.gates  = (float*)take((size_t)3*BB*NCH, 4);
  p.hq     = (__bf16*)take((size_t)BB*CC*HH, 2);
  p.hk     = (__bf16*)take((size_t)BB*CC*HH, 2);
  p.hkT    = (__bf16*)take((size_t)BB*HH*CC, 2);
  p.gate   = (float*)take((size_t)BB*CC*HH, 4);
  p.dvp    = (__bf16*)take((size_t)BB*CC*DD, 2);
  p.dvpT   = (__bf16*)take((size_t)BB*DD*CC, 2);
  p.dhT    = (__bf16*)take((size_t)BB*HH*CC, 2);
  (void)ws_size; (void)in_sizes; (void)n_in; (void)out_size;

  k_cvt  <<<(DD*DD+255)/256, 256, 0, stream>>>(p);
  k_init <<<dim3((HH*DD+255)/256, BB), 256, 0, stream>>>(p);
  k_gates<<<BB*NCH, 64, 0, stream>>>(p);
  k_qkvn <<<BB*(SS/16), 256, 0, stream>>>(p);

  for (int ci = 0; ci < NCH; ++ci) {
    kA <<<BB*16,  256, 0, stream>>>(p, ci);
    kB <<<BB*16,  256, 0, stream>>>(p, ci);
    kC1<<<BB*16,  256, 0, stream>>>(p, ci);
    kC2<<<BB*128, 256, 0, stream>>>(p, ci);
    kD <<<BB*128, 256, 0, stream>>>(p, ci);
    kUb<<<BB,     512, 0, stream>>>(p, ci);
  }
}
